// ConvGPTAttention_83167746720346
// MI455X (gfx1250) — matrix operand order repaired and checked by reading, not compiled, not run on hardware
//
#include <hip/hip_runtime.h>
#include <hip/hip_bf16.h>
#include <stdint.h>

typedef __bf16 bf16;
typedef __attribute__((ext_vector_type(16))) __bf16 v16bf;
typedef __attribute__((ext_vector_type(8)))  float  v8f;
typedef int v4i __attribute__((__vector_size__(16)));

#define T_TOK 4096
#define S_SEQ 2048
#define B_BATCH 2
#define H_HEADS 16
#define KV_HEADS 8
#define D_HEAD 128
#define HID 2048
#define NQKV 4096   // H*D + 2*KV*D
#define SCALE_ATTN 0.08838834764831845f

// CDNA5 async global->LDS copy (ASYNCcnt path), if this toolchain exposes it.
#if defined(__has_builtin)
#if __has_builtin(__builtin_amdgcn_global_load_async_to_lds_b128) && \
    __has_builtin(__builtin_amdgcn_s_wait_asynccnt)
#define HAVE_ASYNC_LDS 1
#endif
#endif

#ifdef HAVE_ASYNC_LDS
__device__ __forceinline__ void async_cp16(const bf16* g, bf16* l) {
  __builtin_amdgcn_global_load_async_to_lds_b128(
      (__attribute__((address_space(1))) v4i*)g,
      (__attribute__((address_space(3))) v4i*)l, 0, 0);
}
#endif

// ---------------------------------------------------------------------------
// WMMA helper: D = A(16x32 bf16) * B(32x16 bf16) + C(16x16 f32)
// ---------------------------------------------------------------------------
__device__ __forceinline__ v8f wmma_bf16(v16bf a, v16bf b, v8f c) {
  return __builtin_amdgcn_wmma_f32_16x16x32_bf16(
      /*neg_a=*/false, a, /*neg_b=*/false, b,
      /*c_mod=*/(short)0, c, /*reuse_a=*/false, /*reuse_b=*/false);
}

// A-fragment (16x32, 16-bit): lane&15 = M row; lane half 0 holds
// K = {0..7, 16..23}, lane half 1 holds K = {8..15, 24..31}.
__device__ __forceinline__ v16bf ldfragA(const bf16* base, int ld, int lane) {
  const bf16* p = base + (size_t)(lane & 15) * ld + ((lane >> 4) << 3);
  union { v16bf v; uint4 u[2]; } r;
  r.u[0] = *(const uint4*)p;
  r.u[1] = *(const uint4*)(p + 16);
  return r.v;
}

// B-fragment (32x16, 16-bit): lane&15 = N column; lane half 0 holds
// K = {0..7, 16..23}, lane half 1 holds K = {8..15, 24..31}.
__device__ __forceinline__ v16bf ldfragB(const bf16* base, int ld, int lane) {
  const bf16* p = base + (size_t)(lane & 15) * ld + ((lane >> 4) << 3);
  union { v16bf v; uint4 u[2]; } r;
  r.u[0] = *(const uint4*)p;
  r.u[1] = *(const uint4*)(p + 16);
  return r.v;
}

// ---------------------------------------------------------------------------
// fp32 -> bf16 convert
// ---------------------------------------------------------------------------
__global__ void cvt_bf16_kernel(const float* __restrict__ src,
                                bf16* __restrict__ dst, int n) {
  int i = blockIdx.x * blockDim.x + threadIdx.x;
  if (i < n) dst[i] = (bf16)src[i];
}

// ---------------------------------------------------------------------------
// C(M,N) = A(M,K) * B(N,K)^T    A,B bf16 row-major, C f32 row-major.
// Block tile 128x128, BK=32, 256 threads = 8 waves; wave -> 32x64 subtile.
// Staging: async global->LDS double-buffer if available, else register
// pipelined b128 copies. Requires M%128==0, N%128==0, K%32==0.
// ---------------------------------------------------------------------------
__global__ __launch_bounds__(256)
void gemm_bf16_nt_kernel(const bf16* __restrict__ A, const bf16* __restrict__ B,
                         float* __restrict__ C, int M, int N, int K) {
  const int tid = threadIdx.x;
  const int lane = tid & 31;
  const int wave = tid >> 5;
  const int wm = (wave & 3) * 32;   // 4 waves over 128 rows
  const int wn = (wave >> 2) * 64;  // 2 waves over 128 cols
  const int m0 = blockIdx.y * 128;
  const int n0 = blockIdx.x * 128;

  // staging coords: tile = 128 rows x 32 elems (64B, 4 x 16B chunks per row).
  // 512 chunks / 256 threads -> rows (ra, ra+64) at 8-elem column ca.
  const int ra = tid >> 2;
  const int ca = (tid & 3) * 8;

  const bf16* Apr0 = A + (size_t)(m0 + ra) * K + ca;
  const bf16* Apr1 = A + (size_t)(m0 + ra + 64) * K + ca;
  const bf16* Bpr0 = B + (size_t)(n0 + ra) * K + ca;
  const bf16* Bpr1 = B + (size_t)(n0 + ra + 64) * K + ca;

  v8f acc[2][4] = {};

#ifdef HAVE_ASYNC_LDS
  // ---- async DMA double-buffered staging ---------------------------------
  __shared__ bf16 As[2][128 * 32];  // 2 x 8 KB
  __shared__ bf16 Bs[2][128 * 32];  // 2 x 8 KB

  // stage tile k0=0 into buffer 0
  async_cp16(Apr0, &As[0][ra * 32 + ca]);
  async_cp16(Apr1, &As[0][(ra + 64) * 32 + ca]);
  async_cp16(Bpr0, &Bs[0][ra * 32 + ca]);
  async_cp16(Bpr1, &Bs[0][(ra + 64) * 32 + ca]);

  int cur = 0;
  for (int k0 = 0; k0 < K; k0 += 32) {
    __builtin_amdgcn_s_wait_asynccnt(0);
    __syncthreads();  // current buffer fully staged by all waves

    if (k0 + 32 < K) {  // DMA next tile into the other buffer during compute
      int nxt = cur ^ 1;
      async_cp16(Apr0 + k0 + 32, &As[nxt][ra * 32 + ca]);
      async_cp16(Apr1 + k0 + 32, &As[nxt][(ra + 64) * 32 + ca]);
      async_cp16(Bpr0 + k0 + 32, &Bs[nxt][ra * 32 + ca]);
      async_cp16(Bpr1 + k0 + 32, &Bs[nxt][(ra + 64) * 32 + ca]);
    }

    const bf16* as = As[cur];
    const bf16* bs = Bs[cur];
    v16bf a0 = ldfragA(as + (wm + 0) * 32, 32, lane);
    v16bf a1 = ldfragA(as + (wm + 16) * 32, 32, lane);
    v16bf b0 = ldfragB(bs + (wn + 0) * 32, 32, lane);
    v16bf b1 = ldfragB(bs + (wn + 16) * 32, 32, lane);
    v16bf b2 = ldfragB(bs + (wn + 32) * 32, 32, lane);
    v16bf b3 = ldfragB(bs + (wn + 48) * 32, 32, lane);

    acc[0][0] = wmma_bf16(a0, b0, acc[0][0]);
    acc[0][1] = wmma_bf16(a0, b1, acc[0][1]);
    acc[0][2] = wmma_bf16(a0, b2, acc[0][2]);
    acc[0][3] = wmma_bf16(a0, b3, acc[0][3]);
    acc[1][0] = wmma_bf16(a1, b0, acc[1][0]);
    acc[1][1] = wmma_bf16(a1, b1, acc[1][1]);
    acc[1][2] = wmma_bf16(a1, b2, acc[1][2]);
    acc[1][3] = wmma_bf16(a1, b3, acc[1][3]);

    cur ^= 1;
  }
#else
  // ---- register-pipelined staging (fallback) -----------------------------
  __shared__ bf16 As[128 * 32];
  __shared__ bf16 Bs[128 * 32];

  uint4* As0 = (uint4*)(As + ra * 32 + ca);
  uint4* As1 = (uint4*)(As + (ra + 64) * 32 + ca);
  uint4* Bs0 = (uint4*)(Bs + ra * 32 + ca);
  uint4* Bs1 = (uint4*)(Bs + (ra + 64) * 32 + ca);

  {
    uint4 a0 = *(const uint4*)Apr0;
    uint4 a1 = *(const uint4*)Apr1;
    uint4 b0 = *(const uint4*)Bpr0;
    uint4 b1 = *(const uint4*)Bpr1;
    *As0 = a0; *As1 = a1; *Bs0 = b0; *Bs1 = b1;
  }
  __syncthreads();

  for (int k0 = 0; k0 < K; k0 += 32) {
    const bool last = (k0 + 32 >= K);
    uint4 na0, na1, nb0, nb1;
    if (!last) {
      na0 = *(const uint4*)(Apr0 + k0 + 32);
      na1 = *(const uint4*)(Apr1 + k0 + 32);
      nb0 = *(const uint4*)(Bpr0 + k0 + 32);
      nb1 = *(const uint4*)(Bpr1 + k0 + 32);
    }

    v16bf a0 = ldfragA(As + (wm + 0) * 32, 32, lane);
    v16bf a1 = ldfragA(As + (wm + 16) * 32, 32, lane);
    v16bf b0 = ldfragB(Bs + (wn + 0) * 32, 32, lane);
    v16bf b1 = ldfragB(Bs + (wn + 16) * 32, 32, lane);
    v16bf b2 = ldfragB(Bs + (wn + 32) * 32, 32, lane);
    v16bf b3 = ldfragB(Bs + (wn + 48) * 32, 32, lane);

    acc[0][0] = wmma_bf16(a0, b0, acc[0][0]);
    acc[0][1] = wmma_bf16(a0, b1, acc[0][1]);
    acc[0][2] = wmma_bf16(a0, b2, acc[0][2]);
    acc[0][3] = wmma_bf16(a0, b3, acc[0][3]);
    acc[1][0] = wmma_bf16(a1, b0, acc[1][0]);
    acc[1][1] = wmma_bf16(a1, b1, acc[1][1]);
    acc[1][2] = wmma_bf16(a1, b2, acc[1][2]);
    acc[1][3] = wmma_bf16(a1, b3, acc[1][3]);

    if (!last) {
      __syncthreads();
      *As0 = na0; *As1 = na1; *Bs0 = nb0; *Bs1 = nb1;
      __syncthreads();
    }
  }
#endif

  // store: C layout — element vv holds M = vv + 8*(lane>=16), N = lane&15
  const int g = lane >> 4;
  const int nn = lane & 15;
#pragma unroll
  for (int i = 0; i < 2; ++i)
#pragma unroll
    for (int j = 0; j < 4; ++j)
#pragma unroll
      for (int vv = 0; vv < 8; ++vv) {
        int row = m0 + wm + i * 16 + vv + 8 * g;
        int col = n0 + wn + j * 16 + nn;
        C[(size_t)row * N + col] = acc[i][j][vv];
      }
}

// ---------------------------------------------------------------------------
// Per (token, head): RMSNorm (q/k only) + RoPE (q/k only) + bf16 pack.
// grid = (T, 32): head 0..15 -> q, 16..23 -> k, 24..31 -> v. block = 128 = D.
// ---------------------------------------------------------------------------
__global__ void norm_rope_kernel(const float* __restrict__ qkv,
                                 const int* __restrict__ positions,
                                 const float* __restrict__ qw,
                                 const float* __restrict__ kw,
                                 bf16* __restrict__ qo, bf16* __restrict__ ko,
                                 bf16* __restrict__ vo) {
  const int t = blockIdx.x;
  const int head = blockIdx.y;
  const int d = threadIdx.x;
  const float* row = qkv + (size_t)t * NQKV + head * D_HEAD;
  float x = row[d];

  if (head >= 24) {  // v head: straight convert
    vo[((size_t)t * KV_HEADS + (head - 24)) * D_HEAD + d] = (bf16)x;
    return;
  }

  __shared__ float sh[D_HEAD];
  __shared__ float red[4];

  float ss = x * x;
#pragma unroll
  for (int off = 16; off > 0; off >>= 1) ss += __shfl_down(ss, off);
  if ((d & 31) == 0) red[d >> 5] = ss;
  __syncthreads();
  float tot = red[0] + red[1] + red[2] + red[3];
  float inv = rsqrtf(tot * (1.0f / D_HEAD) + 1e-6f);

  const float* w = (head < H_HEADS) ? qw : kw;
  sh[d] = x * inv * w[d];
  __syncthreads();

  const int pos = positions[t];
  const int j = d & 63;
  float freq = __powf(10000.0f, -(float)j * (1.0f / 64.0f));
  float ang = (float)pos * freq;
  float c = __cosf(ang), s = __sinf(ang);
  float out = (d < 64) ? (sh[d] * c - sh[d + 64] * s)
                       : (sh[d] * c + sh[d - 64] * s);

  if (head < H_HEADS)
    qo[((size_t)t * H_HEADS + head) * D_HEAD + d] = (bf16)out;
  else
    ko[((size_t)t * KV_HEADS + (head - H_HEADS)) * D_HEAD + d] = (bf16)out;
}

// ---------------------------------------------------------------------------
// Flash-style causal attention, one wave (32 threads) per 16-query tile.
// grid = (S/16, H, B). Scores computed transposed (S^T = K * Q^T) so each
// lane owns one query column -> softmax stats reduce via shfl_xor(16).
// Key tiles of 32 (K-dim of the P*V WMMA). q/k/v bf16 [t][head][d].
// ---------------------------------------------------------------------------
__global__ __launch_bounds__(32)
void flash_attn_kernel(const bf16* __restrict__ q, const bf16* __restrict__ k,
                       const bf16* __restrict__ v, bf16* __restrict__ o) {
  const int qt = blockIdx.x;
  const int h = blockIdx.y;
  const int b = blockIdx.z;
  const int lane = threadIdx.x;
  const int kvh = h >> 1;  // GQA: H/KV = 2
  const int q0 = qt * 16;
  const int ldq = H_HEADS * D_HEAD;
  const int ldk = KV_HEADS * D_HEAD;

  __shared__ bf16 VT[D_HEAD * 32];   // [d][key]  8 KB
  __shared__ bf16 P[16 * 32];        // [query][key] 1 KB

  const size_t qbase = ((size_t)(b * S_SEQ + q0) * H_HEADS + h) * D_HEAD;
  v16bf qfrag[4];
#pragma unroll
  for (int c = 0; c < 4; ++c) qfrag[c] = ldfragB(q + qbase + c * 32, ldq, lane);

  const int g = lane >> 4;
  const int qi = lane & 15;
  const int tq = q0 + qi;

  v8f out[8] = {};
  float m = -3.0e38f, l = 0.0f;

  const int nkt = (q0 + 16 + 31) >> 5;
  for (int kt = 0; kt < nkt; ++kt) {
    const int kt0 = kt * 32;

    // stage V tile transposed: lane = key row; b128 global loads, b16 scatter
    {
      const bf16* vrow =
          v + ((size_t)(b * S_SEQ + kt0 + lane) * KV_HEADS + kvh) * D_HEAD;
      const uint4* v4 = (const uint4*)vrow;
#pragma unroll
      for (int c8 = 0; c8 < 16; ++c8) {
        union { uint4 u; bf16 e[8]; } tmp;
        tmp.u = v4[c8];
#pragma unroll
        for (int e = 0; e < 8; ++e) VT[(c8 * 8 + e) * 32 + lane] = tmp.e[e];
      }
    }

    // scores^T: two 16(key)x16(query) frags, K-dim = d, 8 WMMAs
    const size_t kbase = ((size_t)(b * S_SEQ + kt0) * KV_HEADS + kvh) * D_HEAD;
    v8f st0 = {}, st1 = {};
#pragma unroll
    for (int c = 0; c < 4; ++c) {
      v16bf ka0 = ldfragA(k + kbase + c * 32, ldk, lane);
      v16bf ka1 = ldfragA(k + kbase + (size_t)16 * ldk + c * 32, ldk, lane);
      st0 = wmma_bf16(ka0, qfrag[c], st0);
      st1 = wmma_bf16(ka1, qfrag[c], st1);
    }

    // mask + online softmax (this lane: 16 key-scores for query qi)
    float sc[16];
    float mt = -3.0e38f;
#pragma unroll
    for (int j = 0; j < 2; ++j)
#pragma unroll
      for (int vv = 0; vv < 8; ++vv) {
        int key = kt0 + j * 16 + 8 * g + vv;
        float sv = (j ? st1[vv] : st0[vv]) * SCALE_ATTN;
        if (key > tq) sv = -3.0e38f;
        sc[j * 8 + vv] = sv;
        mt = fmaxf(mt, sv);
      }
    mt = fmaxf(mt, __shfl_xor(mt, 16));
    float mnew = fmaxf(m, mt);
    float alpha = __expf(m - mnew);
    m = mnew;

    float lsum = 0.0f;
#pragma unroll
    for (int j = 0; j < 2; ++j)
#pragma unroll
      for (int vv = 0; vv < 8; ++vv) {
        float p = __expf(sc[j * 8 + vv] - mnew);
        lsum += p;
        P[qi * 32 + j * 16 + 8 * g + vv] = (bf16)p;
      }
    lsum += __shfl_xor(lsum, 16);
    l = l * alpha + lsum;

    float ab[8];
#pragma unroll
    for (int vv = 0; vv < 8; ++vv) ab[vv] = __shfl(alpha, 8 * g + vv);
#pragma unroll
    for (int f = 0; f < 8; ++f)
#pragma unroll
      for (int vv = 0; vv < 8; ++vv) out[f][vv] *= ab[vv];

    __syncthreads();

    // out += P(16q x 32key) * V(32key x 128d): 8 WMMAs
    v16bf pa = ldfragA(P, 32, lane);
#pragma unroll
    for (int f = 0; f < 8; ++f) {
      v16bf vbf = ldfragB(VT + f * 16 * 32, 32, lane);
      out[f] = wmma_bf16(pa, vbf, out[f]);
    }
    __syncthreads();
  }

  float linv[8];
#pragma unroll
  for (int vv = 0; vv < 8; ++vv) linv[vv] = 1.0f / __shfl(l, 8 * g + vv);
  const int nn = lane & 15;
#pragma unroll
  for (int f = 0; f < 8; ++f)
#pragma unroll
    for (int vv = 0; vv < 8; ++vv) {
      int qrow = vv + 8 * g;
      size_t idx =
          ((size_t)(b * S_SEQ + q0 + qrow) * H_HEADS + h) * D_HEAD + f * 16 + nn;
      o[idx] = (bf16)(out[f][vv] * linv[vv]);
    }
}

// ---------------------------------------------------------------------------
// launcher
// ---------------------------------------------------------------------------
extern "C" void kernel_launch(void* const* d_in, const int* in_sizes, int n_in,
                              void* d_out, int out_size, void* d_ws, size_t ws_size,
                              hipStream_t stream) {
  const float* hs    = (const float*)d_in[0];
  const int*   pos   = (const int*)d_in[1];
  const float* w_qkv = (const float*)d_in[2];
  const float* w_o   = (const float*)d_in[3];
  const float* qnw   = (const float*)d_in[4];
  const float* knw   = (const float*)d_in[5];
  float* out = (float*)d_out;

  char* w = (char*)d_ws;
  auto alloc = [&](size_t bytes) {
    char* p = w;
    w += (bytes + 255) & ~(size_t)255;
    return p;
  };
  bf16*  hs_b   = (bf16*)alloc((size_t)T_TOK * HID * 2);
  bf16*  wqkv_b = (bf16*)alloc((size_t)NQKV * HID * 2);
  bf16*  wo_b   = (bf16*)alloc((size_t)HID * HID * 2);
  float* qkv    = (float*)alloc((size_t)T_TOK * NQKV * 4);
  bf16*  q_b    = (bf16*)alloc((size_t)T_TOK * H_HEADS * D_HEAD * 2);
  bf16*  k_b    = (bf16*)alloc((size_t)T_TOK * KV_HEADS * D_HEAD * 2);
  bf16*  v_b    = (bf16*)alloc((size_t)T_TOK * KV_HEADS * D_HEAD * 2);
  bf16*  attn_b = (bf16*)alloc((size_t)T_TOK * H_HEADS * D_HEAD * 2);

  {
    int n = T_TOK * HID;
    cvt_bf16_kernel<<<(n + 255) / 256, 256, 0, stream>>>(hs, hs_b, n);
    n = NQKV * HID;
    cvt_bf16_kernel<<<(n + 255) / 256, 256, 0, stream>>>(w_qkv, wqkv_b, n);
    n = HID * HID;
    cvt_bf16_kernel<<<(n + 255) / 256, 256, 0, stream>>>(w_o, wo_b, n);
  }
  // qkv = hs @ w_qkv^T   (4096 x 4096 x 2048)
  gemm_bf16_nt_kernel<<<dim3(NQKV / 128, T_TOK / 128), 256, 0, stream>>>(
      hs_b, wqkv_b, qkv, T_TOK, NQKV, HID);
  // rmsnorm + rope + pack
  norm_rope_kernel<<<dim3(T_TOK, 32), 128, 0, stream>>>(
      qkv, pos, qnw, knw, q_b, k_b, v_b);
  // causal attention
  flash_attn_kernel<<<dim3(S_SEQ / 16, H_HEADS, B_BATCH), 32, 0, stream>>>(
      q_b, k_b, v_b, attn_b);
  // out = attn @ w_o^T   (4096 x 2048 x 2048)
  gemm_bf16_nt_kernel<<<dim3(HID / 128, T_TOK / 128), 256, 0, stream>>>(
      attn_b, wo_b, out, T_TOK, HID, HID);
}
